// EnergyPriceLSTM_18665927868932
// MI455X (gfx1250) — compile-verified, matrix operand order checked by reading, not run on hardware
//
#include <hip/hip_runtime.h>
#include <hip/hip_bf16.h>

typedef __attribute__((ext_vector_type(16))) _Float16 v16h;
typedef __attribute__((ext_vector_type(8)))  float    v8f;

#define T_STEPS   168
#define INPUT_DIM 19
#define HID       64

// Hardware reciprocal: single v_rcp_f32 (TRANS pipe) instead of the
// correctly-rounded v_div_scale/v_div_fmas/v_div_fixup sequence.
__device__ __forceinline__ float fast_rcp(float x) {
    return __builtin_amdgcn_rcpf(x);
}
__device__ __forceinline__ float fast_sigmoid(float x) {
    // v_exp_f32 + v_rcp_f32, both TRANS ops that co-execute with WMMA
    return fast_rcp(1.0f + __expf(-x));
}
__device__ __forceinline__ float fast_tanh(float x) {
#if __has_builtin(__builtin_amdgcn_tanhf)
    return __builtin_amdgcn_tanhf(x);       // native v_tanh_f32 on gfx1250
#else
    return 2.0f * fast_rcp(1.0f + __expf(-2.0f * x)) - 1.0f;
#endif
}

// K index of the low half of packed pair p (0..7) for lane-group g (0/1),
// per CDNA5 16-bit fragment layout (CDNA5 ISA 7.12.2):
//   VGPR0..3: K = 2p + 8g ; VGPR4..7: K = 16 + 2(p-4) + 8g
__device__ __forceinline__ int kpair_base(int p, int g) {
    return (p < 4) ? (2 * p + 8 * g) : (16 + 2 * (p - 4) + 8 * g);
}

template <typename F>
__device__ __forceinline__ v16h make_frag(F f, int g) {
    v16h r;
#pragma unroll
    for (int p = 0; p < 8; ++p) {
        int kb = kpair_base(p, g);
        r[2 * p]     = (_Float16)f(kb);
        r[2 * p + 1] = (_Float16)f(kb + 1);
    }
    return r;
}

// Load a 16x32 f16 A-fragment from LDS (row-major, rowStride in halves) at
// column offset colH (halves). Two aligned 16B reads/lane -> 2x ds_load_b128.
__device__ __forceinline__ v16h lds_afrag(const _Float16* buf, int rowStride,
                                          int colH, int m, int g) {
    const _Float16* p0 = buf + m * rowStride + colH + g * 8;        // K =  8g .. 8g+7
    const _Float16* p1 = buf + m * rowStride + colH + 16 + g * 8;   // K = 16+8g .. 16+8g+7
    v16h r;
#pragma unroll
    for (int i = 0; i < 8; ++i) { r[i] = p0[i]; r[8 + i] = p1[i]; }
    return r;
}

__global__ __launch_bounds__(128)
void lstm_fused_kernel(const float* __restrict__ x,
                       const float* __restrict__ Wih0, const float* __restrict__ Whh0,
                       const float* __restrict__ bih0, const float* __restrict__ bhh0,
                       const float* __restrict__ Wih1, const float* __restrict__ Whh1,
                       const float* __restrict__ bih1, const float* __restrict__ bhh1,
                       const float* __restrict__ Wfc,  const float* __restrict__ bfc,
                       float* __restrict__ out, int B)
{
    // Cross-wave h exchange buffers (16 batch rows x 64 hidden, f16) — 4KB total
    __shared__ __attribute__((aligned(16))) _Float16 haBuf[16 * HID];
    __shared__ __attribute__((aligned(16))) _Float16 hbBuf[16 * HID];

    const int tid  = threadIdx.x;
    const int w    = tid >> 5;      // wave id 0..3 (wave32)
    const int lane = tid & 31;
    const int nl   = lane & 15;     // M (A) / N (B,C,D) position within 16x16 tile
    const int g    = lane >> 4;     // lane group

    for (int i = tid; i < 16 * HID; i += 128) {
        haBuf[i] = (_Float16)0.0f;
        hbBuf[i] = (_Float16)0.0f;
    }

    const int brow = blockIdx.x * 16 + nl;
    const int b    = (brow < B) ? brow : (B - 1);

    // ---- one-time: weight B-fragments into registers (f16), biases folded ----
    v16h  w0f[4][3];   // layer0: gates x {x-frag, h-frag0, h-frag1}
    v16h  w1f[4][4];   // layer1: gates x {hA0, hA1, hB0, hB1}
    float bias0[4], bias1[4];
#pragma unroll
    for (int gt = 0; gt < 4; ++gt) {
        const int n = gt * 64 + w * 16 + nl;   // gate-output row this lane produces
        bias0[gt] = bih0[n] + bhh0[n];
        bias1[gt] = bih1[n] + bhh1[n];
#pragma unroll
        for (int kf = 0; kf < 3; ++kf) {
            w0f[gt][kf] = make_frag([&](int kk) -> float {
                int k = kf * 32 + kk;
                if (k < 32) return (k < INPUT_DIM) ? Wih0[n * INPUT_DIM + k] : 0.0f;
                return Whh0[n * HID + (k - 32)];
            }, g);
        }
#pragma unroll
        for (int kf = 0; kf < 4; ++kf) {
            w1f[gt][kf] = make_frag([&](int kk) -> float {
                int k = kf * 32 + kk;
                return (k < HID) ? Wih1[n * HID + k] : Whh1[n * HID + (k - HID)];
            }, g);
        }
    }

    v8f ca, cb;   // persistent cell state, one 16x16 slice per wave per layer
#pragma unroll
    for (int r = 0; r < 8; ++r) { ca[r] = 0.0f; cb[r] = 0.0f; }

    __syncthreads();

    const int hcol = w * 16 + nl;   // hidden column this wave owns

    for (int t = 0; t < T_STEPS; ++t) {
        // ================= layer 0 =================
        const float* xp = x + ((long)b * T_STEPS + t) * INPUT_DIM;
        if (t + 1 < T_STEPS) __builtin_prefetch(xp + INPUT_DIM, 0, 0);
        v16h a0x = make_frag([&](int kk) -> float {
            return (kk < INPUT_DIM) ? xp[kk] : 0.0f;      // pad K 19->32
        }, g);
        v16h a0h0 = lds_afrag(haBuf, HID, 0,  nl, g);
        v16h a0h1 = lds_afrag(haBuf, HID, 32, nl, g);
        __syncthreads();   // WAR: done reading old h before rewriting

        v8f z[4];
#pragma unroll
        for (int gt = 0; gt < 4; ++gt) {
            v8f acc;
#pragma unroll
            for (int r = 0; r < 8; ++r) acc[r] = bias0[gt];
            acc = __builtin_amdgcn_wmma_f32_16x16x32_f16(false, a0x,  false, w0f[gt][0], (short)0, acc, false, false);
            acc = __builtin_amdgcn_wmma_f32_16x16x32_f16(false, a0h0, false, w0f[gt][1], (short)0, acc, false, false);
            acc = __builtin_amdgcn_wmma_f32_16x16x32_f16(false, a0h1, false, w0f[gt][2], (short)0, acc, false, false);
            z[gt] = acc;
        }
#pragma unroll
        for (int r = 0; r < 8; ++r) {            // torch gate order i,f,g,o
            float ig = fast_sigmoid(z[0][r]);
            float fg = fast_sigmoid(z[1][r]);
            float gg = fast_tanh(z[2][r]);
            float og = fast_sigmoid(z[3][r]);
            float c  = fg * ca[r] + ig * gg;
            ca[r] = c;
            float h = og * fast_tanh(c);
            haBuf[(r + 8 * g) * HID + hcol] = (_Float16)h;   // D-layout: M=r+8g, N=nl
        }
        __syncthreads();   // RAW: new hA visible to all waves

        // ================= layer 1 =================
        v16h a1[4];
        a1[0] = lds_afrag(haBuf, HID, 0,  nl, g);
        a1[1] = lds_afrag(haBuf, HID, 32, nl, g);
        a1[2] = lds_afrag(hbBuf, HID, 0,  nl, g);
        a1[3] = lds_afrag(hbBuf, HID, 32, nl, g);
        __syncthreads();   // WAR: done reading old hB

#pragma unroll
        for (int gt = 0; gt < 4; ++gt) {
            v8f acc;
#pragma unroll
            for (int r = 0; r < 8; ++r) acc[r] = bias1[gt];
#pragma unroll
            for (int kf = 0; kf < 4; ++kf)
                acc = __builtin_amdgcn_wmma_f32_16x16x32_f16(false, a1[kf], false, w1f[gt][kf], (short)0, acc, false, false);
            z[gt] = acc;
        }
#pragma unroll
        for (int r = 0; r < 8; ++r) {
            float ig = fast_sigmoid(z[0][r]);
            float fg = fast_sigmoid(z[1][r]);
            float gg = fast_tanh(z[2][r]);
            float og = fast_sigmoid(z[3][r]);
            float c  = fg * cb[r] + ig * gg;
            cb[r] = c;
            float h = og * fast_tanh(c);
            hbBuf[(r + 8 * g) * HID + hcol] = (_Float16)h;
        }
        __syncthreads();   // RAW: new hB visible for next step / final dot
    }

    // ---- final projection: out[b] = hB[T-1] . Wfc + bfc ----
    if (tid < 16) {
        int ob = blockIdx.x * 16 + tid;
        if (ob < B) {
            float s = bfc[0];
#pragma unroll
            for (int k = 0; k < HID; ++k)
                s += (float)hbBuf[tid * HID + k] * Wfc[k];
            out[ob] = s;
        }
    }
}

extern "C" void kernel_launch(void* const* d_in, const int* in_sizes, int n_in,
                              void* d_out, int out_size, void* d_ws, size_t ws_size,
                              hipStream_t stream) {
    const float* x    = (const float*)d_in[0];
    const float* Wih0 = (const float*)d_in[1];
    const float* Whh0 = (const float*)d_in[2];
    const float* bih0 = (const float*)d_in[3];
    const float* bhh0 = (const float*)d_in[4];
    const float* Wih1 = (const float*)d_in[5];
    const float* Whh1 = (const float*)d_in[6];
    const float* bih1 = (const float*)d_in[7];
    const float* bhh1 = (const float*)d_in[8];
    const float* Wfc  = (const float*)d_in[9];
    const float* bfc  = (const float*)d_in[10];
    float* out = (float*)d_out;

    int B = in_sizes[0] / (T_STEPS * INPUT_DIM);   // 4096
    int grid = (B + 15) / 16;                      // 16 batch rows per block
    lstm_fused_kernel<<<grid, 128, 0, stream>>>(x, Wih0, Whh0, bih0, bhh0,
                                                Wih1, Whh1, bih1, bhh1,
                                                Wfc, bfc, out, B);
}
